// MultiHeadAttention_14181982011646
// MI455X (gfx1250) — matrix operand order repaired and checked by reading, not compiled, not run on hardware
//
#include <hip/hip_runtime.h>
#include <hip/hip_bf16.h>

#define Bsz 4
#define Sdim 2048
#define Hdim 512
#define NHd 8
#define Dd 64
#define Kdim Hdim          // GEMM K for projections = 512
#define Mrows (Bsz * Sdim) // 8192

typedef _Float16 h16 __attribute__((ext_vector_type(16)));
typedef _Float16 h8  __attribute__((ext_vector_type(8)));
typedef float    f8  __attribute__((ext_vector_type(8)));

union H16 { h16 v; h8 h[2]; };

__device__ __forceinline__ f8 wmma_f16(h16 a, h16 b, f8 c) {
  // (neg_a, A, neg_b, B, c_mod, C, reuse_a, reuse_b)
  return __builtin_amdgcn_wmma_f32_16x16x32_f16(false, a, false, b, (short)0, c,
                                                false, false);
}

// --- CDNA5 async global->LDS copy (ASYNCcnt) with safe fallback -------------
#if __has_builtin(__builtin_amdgcn_global_load_async_to_lds_b128) && \
    __has_builtin(__builtin_amdgcn_s_wait_asynccnt)
#define USE_ASYNC_LDS 1
#else
#define USE_ASYNC_LDS 0
#endif

#if USE_ASYNC_LDS
typedef int gv4i __attribute__((vector_size(16)));
typedef __attribute__((address_space(1))) gv4i* gv4i_gptr; // global int4*
typedef __attribute__((address_space(3))) gv4i* gv4i_lptr; // LDS int4*
#endif

__device__ __forceinline__ void copy16B_to_lds(const _Float16* g, _Float16* l) {
#if USE_ASYNC_LDS
  __builtin_amdgcn_global_load_async_to_lds_b128((gv4i_gptr)g, (gv4i_lptr)l, 0,
                                                 0);
#else
  *(uint4*)l = *(const uint4*)g;
#endif
}

__device__ __forceinline__ void wait_lds_copies() {
#if USE_ASYNC_LDS
  __builtin_amdgcn_s_wait_asynccnt(0);
#endif
}

// ---------------------------------------------------------------------------
// Kernel 1: Y = x @ W^T  (three weights via blockIdx.z), f16 output.
// Q, K stored head-major [B, NH, S, D]; V stored TRANSPOSED [B, NH, D, S].
// Block = 128 threads = 4 waves; each block computes a 64x64 tile of Y,
// marching K in 64-wide chunks staged in LDS as f16.
// ---------------------------------------------------------------------------
#define LXS 72 // padded LDS row stride (f16 elems) for 64-wide K chunk

__global__ __launch_bounds__(128) void qkv_proj_kernel(
    const float* __restrict__ x, const float* __restrict__ Wq,
    const float* __restrict__ Wk, const float* __restrict__ Wv,
    _Float16* __restrict__ ws_qkv) {
  __shared__ _Float16 lx[64 * LXS];
  __shared__ _Float16 lw[64 * LXS];

  const int m0 = blockIdx.x * 64;
  const int n0 = blockIdx.y * 64;
  const int wsel = blockIdx.z;
  const float* W = (wsel == 0) ? Wq : ((wsel == 1) ? Wk : Wv);
  _Float16* dst = ws_qkv + (size_t)wsel * (size_t)Bsz * NHd * Sdim * Dd;

  const int tid = threadIdx.x;
  const int wave = tid >> 5;
  const int lane = tid & 31;
  const int half = lane >> 4;
  const int l16 = lane & 15;

  f8 acc[4];
#pragma unroll
  for (int nt = 0; nt < 4; ++nt) acc[nt] = {};

  for (int kc = 0; kc < Kdim; kc += 64) {
    __syncthreads();
    // Stage x tile (64 m x 64 k) and W tile (64 n x 64 k) as f16.
#pragma unroll
    for (int r = 0; r < 4; ++r) {
      const int idx = tid + r * 128;      // 512 chunks of 8 elems per tile
      const int row = idx >> 3;           // 8 chunks per 64-wide row
      const int cb = (idx & 7) * 8;
      {
        const float4* s4 = (const float4*)(x + (size_t)(m0 + row) * Kdim + kc + cb);
        float4 f0 = s4[0], f1 = s4[1];
        h8 hv = {(_Float16)f0.x, (_Float16)f0.y, (_Float16)f0.z, (_Float16)f0.w,
                 (_Float16)f1.x, (_Float16)f1.y, (_Float16)f1.z, (_Float16)f1.w};
        *(h8*)(lx + row * LXS + cb) = hv;
      }
      {
        const float4* s4 = (const float4*)(W + (size_t)(n0 + row) * Kdim + kc + cb);
        float4 f0 = s4[0], f1 = s4[1];
        h8 hv = {(_Float16)f0.x, (_Float16)f0.y, (_Float16)f0.z, (_Float16)f0.w,
                 (_Float16)f1.x, (_Float16)f1.y, (_Float16)f1.z, (_Float16)f1.w};
        *(h8*)(lw + row * LXS + cb) = hv;
      }
    }
    if (kc + 64 < Kdim) // prefetch next x chunk -> global_prefetch_b8
      __builtin_prefetch(x + (size_t)(m0 + (tid >> 3)) * Kdim + kc + 64, 0, 1);
    __syncthreads();

    // A fragments: lane holds row (wave*16 + l16); K blocks {8h, 16+8h}+32*kk
    H16 au[2];
    const _Float16* ar = lx + (wave * 16 + l16) * LXS;
#pragma unroll
    for (int kk = 0; kk < 2; ++kk) {
      au[kk].h[0] = *(const h8*)(ar + 32 * kk + 8 * half);
      au[kk].h[1] = *(const h8*)(ar + 32 * kk + 16 + 8 * half);
    }
#pragma unroll
    for (int nt = 0; nt < 4; ++nt) {
      const _Float16* br = lw + (nt * 16 + l16) * LXS + 8 * half;
#pragma unroll
      for (int kk = 0; kk < 2; ++kk) {
        H16 bu;
        bu.h[0] = *(const h8*)(br + 32 * kk);
        bu.h[1] = *(const h8*)(br + 32 * kk + 16);
        acc[nt] = wmma_f16(au[kk].v, bu.v, acc[nt]);
      }
    }
  }

  // Write out f16. C layout: row = r + 8*half, col = l16 (per 16x16 tile).
#pragma unroll
  for (int nt = 0; nt < 4; ++nt) {
#pragma unroll
    for (int r = 0; r < 8; ++r) {
      const int row = m0 + wave * 16 + 8 * half + r; // global M row
      const int n = n0 + nt * 16 + l16;              // global N col
      const int bb = row >> 11;                      // row / S
      const int s = row & (Sdim - 1);
      const int nh = n >> 6; // n / D
      const int d = n & (Dd - 1);
      const _Float16 v = (_Float16)acc[nt][r];
      if (wsel == 2) // V: transposed layout [B, NH, D, S]
        dst[((size_t)(bb * NHd + nh) * Dd + d) * Sdim + s] = v;
      else           // Q, K: head-major [B, NH, S, D]
        dst[((size_t)(bb * NHd + nh) * Sdim + s) * Dd + d] = v;
    }
  }
}

// ---------------------------------------------------------------------------
// Kernel 2: causal flash attention. Q,K head-major [S,D] f16; V transposed
// [D,S] f16; f32 output. Block = 128 threads = 4 waves; block owns 64 query
// rows of one (b, h).
// ---------------------------------------------------------------------------
#define KSTR 72 // padded LDS strides
#define VSTR 72
#define PSTR 72

__global__ __launch_bounds__(128) void attn_kernel(
    const _Float16* __restrict__ Qg, const _Float16* __restrict__ Kg,
    const _Float16* __restrict__ Vtg, const unsigned char* __restrict__ pmask,
    float* __restrict__ out) {
  __shared__ _Float16 ldsK[64 * KSTR];     // K tile, row-major [key][d]
  __shared__ _Float16 ldsVt[64 * VSTR];    // V tile transposed [d][key]
  __shared__ _Float16 ldsP[4 * 16 * PSTR]; // per-wave P staging
  __shared__ unsigned char ldsM[64];       // padding mask chunk

  const int qt = blockIdx.x;
  const int h = blockIdx.y;
  const int b = blockIdx.z;
  const int q0 = qt * 64;

  const size_t headoff = (size_t)(b * NHd + h) * Sdim * Dd;
  const _Float16* Qh = Qg + headoff;
  const _Float16* Kh = Kg + headoff;
  const _Float16* Vth = Vtg + headoff; // [D, S]

  const int tid = threadIdx.x;
  const int wave = tid >> 5;
  const int lane = tid & 31;
  const int half = lane >> 4;
  const int l16 = lane & 15;

  // Load Q fragments for this wave's 16 rows (2 K-chunks covering D=64).
  H16 qa[2];
  {
    const _Float16* qrow = Qh + (size_t)(q0 + wave * 16 + l16) * Dd;
#pragma unroll
    for (int kc = 0; kc < 2; ++kc) {
      qa[kc].h[0] = *(const h8*)(qrow + 32 * kc + 8 * half);
      qa[kc].h[1] = *(const h8*)(qrow + 32 * kc + 16 + 8 * half);
    }
  }

  f8 o_acc[4];
#pragma unroll
  for (int dt = 0; dt < 4; ++dt) o_acc[dt] = {};
  float m_r[8], l_r[8];
#pragma unroll
  for (int r = 0; r < 8; ++r) {
    m_r[r] = -__builtin_inff();
    l_r[r] = 0.0f;
  }

  const float scale = 0.125f; // 1/sqrt(64)
  const int nkt = qt + 1;     // causal: only key tiles at or before q tile

  // Per-thread staging assignment: 64 bytes (4 x b128) of each 8KB tile.
  const int srow = tid >> 1;          // 0..63
  const int shb = (tid & 1) * 32;     // 32-f16 half of a 64-wide row

  for (int jt = 0; jt < nkt; ++jt) {
    const int j0 = jt * 64;
    __syncthreads();
    // Stage K rows [key][d] and Vt rows [d][key]; both straight b128 copies.
    {
      const _Float16* ks = Kh + (size_t)(j0 + srow) * Dd + shb;
      _Float16* kd = ldsK + srow * KSTR + shb;
      const _Float16* vs = Vth + (size_t)srow * Sdim + j0 + shb;
      _Float16* vd = ldsVt + srow * VSTR + shb;
#pragma unroll
      for (int c = 0; c < 32; c += 8) {
        copy16B_to_lds(ks + c, kd + c);
        copy16B_to_lds(vs + c, vd + c);
      }
      if (tid < 64) ldsM[tid] = pmask[(size_t)b * Sdim + j0 + tid];
      if (jt + 1 < nkt) { // prefetch next tiles -> global_prefetch_b8
        __builtin_prefetch(Kh + (size_t)(j0 + 64 + srow) * Dd + shb, 0, 1);
        __builtin_prefetch(Vth + (size_t)srow * Sdim + j0 + 64 + shb, 0, 1);
      }
      wait_lds_copies(); // s_wait_asynccnt 0 (async path only)
    }
    __syncthreads();

    // Scores: S = Q K^T for this wave's 16 rows x 64 key cols.
    f8 st[4];
#pragma unroll
    for (int nt = 0; nt < 4; ++nt) {
      f8 acc = {};
      const _Float16* kr = ldsK + (nt * 16 + l16) * KSTR + 8 * half;
#pragma unroll
      for (int kc = 0; kc < 2; ++kc) {
        H16 bu;
        bu.h[0] = *(const h8*)(kr + 32 * kc);
        bu.h[1] = *(const h8*)(kr + 32 * kc + 16);
        acc = wmma_f16(qa[kc].v, bu.v, acc);
      }
      st[nt] = acc;
    }

    // Masking + online softmax. C layout: row = r + 8*half, col = l16.
    bool colpad[4];
    int col_g[4];
#pragma unroll
    for (int nt = 0; nt < 4; ++nt) {
      colpad[nt] = (ldsM[nt * 16 + l16] != 0);
      col_g[nt] = j0 + nt * 16 + l16;
    }
#pragma unroll
    for (int r = 0; r < 8; ++r) {
      const int row_g = q0 + wave * 16 + 8 * half + r;
      float mx = -__builtin_inff();
#pragma unroll
      for (int nt = 0; nt < 4; ++nt) {
        float s = st[nt][r] * scale;
        const bool masked = (col_g[nt] > row_g) || colpad[nt];
        s = masked ? -__builtin_inff() : s;
        st[nt][r] = s;
        mx = fmaxf(mx, s);
      }
#pragma unroll
      for (int o = 1; o < 16; o <<= 1) mx = fmaxf(mx, __shfl_xor(mx, o, 32));

      const float mnew = fmaxf(m_r[r], mx);
      const float corr =
          (m_r[r] == -__builtin_inff()) ? 0.0f : __expf(m_r[r] - mnew);
      float rsum = 0.0f;
#pragma unroll
      for (int nt = 0; nt < 4; ++nt) {
        const float sv = st[nt][r];
        const float pv = (sv == -__builtin_inff()) ? 0.0f : __expf(sv - mnew);
        st[nt][r] = pv;
        rsum += pv;
      }
#pragma unroll
      for (int o = 1; o < 16; o <<= 1) rsum += __shfl_xor(rsum, o, 32);
      l_r[r] = l_r[r] * corr + rsum;
      m_r[r] = mnew;
#pragma unroll
      for (int dt = 0; dt < 4; ++dt) o_acc[dt][r] *= corr;
    }

    // Stage P (f16) to LDS to re-layout C-layout -> A-fragment layout.
    {
      _Float16* pw = ldsP + wave * 16 * PSTR;
#pragma unroll
      for (int r = 0; r < 8; ++r) {
        const int rl = 8 * half + r;
#pragma unroll
        for (int nt = 0; nt < 4; ++nt)
          pw[rl * PSTR + nt * 16 + l16] = (_Float16)st[nt][r];
      }
    }
    __syncthreads();

    // O += P @ V (K-dim = 64 keys, 2 chunks of 32).
    {
      const _Float16* pr = ldsP + wave * 16 * PSTR + l16 * PSTR;
#pragma unroll
      for (int kc = 0; kc < 2; ++kc) {
        H16 au;
        au.h[0] = *(const h8*)(pr + 32 * kc + 8 * half);
        au.h[1] = *(const h8*)(pr + 32 * kc + 16 + 8 * half);
#pragma unroll
        for (int dt = 0; dt < 4; ++dt) {
          const _Float16* vr = ldsVt + (dt * 16 + l16) * VSTR + 32 * kc + 8 * half;
          H16 bu;
          bu.h[0] = *(const h8*)vr;
          bu.h[1] = *(const h8*)(vr + 16);
          o_acc[dt] = wmma_f16(au.v, bu.v, o_acc[dt]);
        }
      }
    }
  }

  // Normalize and write f32 output in [B, S, NH*D] layout.
#pragma unroll
  for (int r = 0; r < 8; ++r) {
    const float inv = 1.0f / l_r[r];
    const int s = q0 + wave * 16 + 8 * half + r;
#pragma unroll
    for (int dt = 0; dt < 4; ++dt) {
      const int d = dt * 16 + l16;
      out[((size_t)b * Sdim + s) * Hdim + h * Dd + d] = o_acc[dt][r] * inv;
    }
  }
}

// ---------------------------------------------------------------------------
extern "C" void kernel_launch(void* const* d_in, const int* in_sizes, int n_in,
                              void* d_out, int out_size, void* d_ws,
                              size_t ws_size, hipStream_t stream) {
  const float* x = (const float*)d_in[0];
  const unsigned char* pm = (const unsigned char*)d_in[1]; // bool mask
  const float* Wq = (const float*)d_in[2];
  const float* Wk = (const float*)d_in[3];
  const float* Wv = (const float*)d_in[4];
  float* out = (float*)d_out;

  _Float16* ws = (_Float16*)d_ws;
  const size_t per = (size_t)Bsz * NHd * Sdim * Dd; // f16 elems per tensor

  dim3 g1(Mrows / 64, Hdim / 64, 3), blk(128);
  qkv_proj_kernel<<<g1, blk, 0, stream>>>(x, Wq, Wk, Wv, ws);

  dim3 g2(Sdim / 64, NHd, Bsz);
  attn_kernel<<<g2, blk, 0, stream>>>(ws, ws + per, ws + 2 * per, pm, out);
}
